// STGCN_49323404427952
// MI455X (gfx1250) — compile-verified, matrix operand order checked by reading, not run on hardware
//
#include <hip/hip_runtime.h>
#include <hip/hip_bf16.h>

typedef __attribute__((ext_vector_type(16))) _Float16 v16h;
typedef __attribute__((ext_vector_type(8)))  float    v8f;

#define WAVES_PER_BLOCK 4

__device__ __forceinline__ float sigmoidf_(float x) {
  return 1.0f / (1.0f + __expf(-x));
}

// ---------------------------------------------------------------------------
// Generic WMMA conv-as-GEMM kernel.
// raw[m, co] = sum_{ci,kk} in[node, ci, t+kk] * W[co, ci, kk], m = node*TOUT + t
// kidx = ci*KW + kk  (K padded to multiple of 32 with zeros)
// GLU:  out[node, co, t] = (raw[co]+b[co]) * sigmoid(raw[co+COUT_G]+b[co+COUT_G])
// else: out[node, co, t] = raw[co]
// WCIN_MAJOR: weight stored as W[ci, co] (for the einsum layers, KW==1)
// ---------------------------------------------------------------------------
template<int CIN, int KW, int COUT_RAW, int TIN, bool GLU, bool WCIN_MAJOR>
__launch_bounds__(WAVES_PER_BLOCK * 32)
__global__ void conv_wmma_kernel(const float* __restrict__ in,
                                 const float* __restrict__ w,
                                 const float* __restrict__ bias,
                                 float* __restrict__ out,
                                 int n_nodes)
{
  constexpr int TOUT   = TIN - KW + 1;
  constexpr int KK     = CIN * KW;
  constexpr int KSTEPS = (KK + 31) / 32;
  constexpr int NTILES = COUT_RAW / 16;
  constexpr int COUT_G = GLU ? COUT_RAW / 2 : COUT_RAW;

  __shared__ __align__(32) _Float16 lds_b[NTILES * KSTEPS * 512];

  const int lane = threadIdx.x & 31;
  const int wave = threadIdx.x >> 5;

  // Stage weights into LDS as WMMA B-fragments (16-bit B 32x16 layout):
  // lane L holds column N = L%16; half j holds K = (j<8?j:j+8) + (L<16?0:8)
  for (int idx = threadIdx.x; idx < NTILES * KSTEPS * 512; idx += blockDim.x) {
    const int j    = idx & 15;
    const int l    = (idx >> 4) & 31;
    const int rest = idx >> 9;
    const int ks   = rest % KSTEPS;
    const int nt   = rest / KSTEPS;
    const int ncol = nt * 16 + (l & 15);
    const int kidx = ks * 32 + (j < 8 ? j : j + 8) + ((l < 16) ? 0 : 8);
    _Float16 v = (_Float16)0.0f;
    if (kidx < KK) {
      const int ci = kidx / KW;
      const int kk = kidx % KW;
      const float wv = WCIN_MAJOR ? w[ci * COUT_RAW + ncol]
                                  : w[(ncol * CIN + ci) * KW + kk];
      v = (_Float16)wv;
    }
    lds_b[idx] = v;
  }
  __syncthreads();

  const long long Mtotal = (long long)n_nodes * TOUT;
  const long long tile   = (long long)blockIdx.x * WAVES_PER_BLOCK + wave;
  const long long m0     = tile * 16;
  if (m0 >= Mtotal) return;   // wave-uniform; EXEC stays all-ones for WMMA

  v8f acc[NTILES] = {};

  // A-fragment row owned by this lane (16-bit A 16x32 layout)
  const long long mrow = m0 + (lane & 15);
  const bool mvalid    = (mrow < Mtotal);
  const long long node = mvalid ? (mrow / TOUT) : 0;
  const int t          = mvalid ? (int)(mrow % TOUT) : 0;
  const float* __restrict__ inrow = in + node * (long long)CIN * TIN;

  for (int ks = 0; ks < KSTEPS; ++ks) {
    v16h a;
#pragma unroll
    for (int j = 0; j < 16; ++j) {
      const int kidx = ks * 32 + (j < 8 ? j : j + 8) + ((lane < 16) ? 0 : 8);
      float v = 0.0f;
      if (mvalid && kidx < KK) {
        const int ci = kidx / KW;
        const int kk = kidx % KW;
        v = inrow[ci * TIN + t + kk];
      }
      a[j] = (_Float16)v;
    }
#pragma unroll
    for (int nt = 0; nt < NTILES; ++nt) {
      const v16h b = *(const v16h*)&lds_b[((nt * KSTEPS + ks) * 32 + lane) * 16];
      acc[nt] = __builtin_amdgcn_wmma_f32_16x16x32_f16(
          false, a, false, b, (short)0, acc[nt], false, false);
    }
  }

  // D layout: lane L owns column N=L%16; VGPR r owns row M = r + (L<16?0:8)
#pragma unroll
  for (int r = 0; r < 8; ++r) {
    const long long m = m0 + r + ((lane < 16) ? 0 : 8);
    if (m >= Mtotal) continue;
    const long long onode = m / TOUT;
    const int ot          = (int)(m % TOUT);
    if constexpr (GLU) {
#pragma unroll
      for (int nt = 0; nt < NTILES / 2; ++nt) {
        const int ch   = nt * 16 + (lane & 15);
        const float av = acc[nt][r] + bias[ch];
        const float gv = acc[nt + NTILES / 2][r] + bias[ch + COUT_G];
        out[(onode * COUT_G + ch) * TOUT + ot] = av * sigmoidf_(gv);
      }
    } else {
#pragma unroll
      for (int nt = 0; nt < NTILES; ++nt) {
        const int ch = nt * 16 + (lane & 15);
        out[(onode * COUT_G + ch) * TOUT + ot] = acc[nt][r];
      }
    }
  }
}

// ------------------------- scalar helper kernels ---------------------------

__global__ void zero_f32(float* p, long long n) {
  long long i = blockIdx.x * (long long)blockDim.x + threadIdx.x;
  if (i < n) p[i] = 0.0f;
}

// conv_glu #1: x (N,2,12), w (128,2,3) -> h1 (N,64,10).  K=6: too small for WMMA.
__global__ void conv1_kernel(const float* __restrict__ x,
                             const float* __restrict__ w,
                             const float* __restrict__ b,
                             float* __restrict__ out, int n_nodes) {
  long long idx = blockIdx.x * (long long)blockDim.x + threadIdx.x;
  long long tot = (long long)n_nodes * 64 * 10;
  if (idx >= tot) return;
  const int t  = (int)(idx % 10);
  const long long r = idx / 10;
  const int co = (int)(r % 64);
  const long long n = r / 64;
  const float* xp = x + n * 24;
  float a = b[co], g = b[co + 64];
#pragma unroll
  for (int ci = 0; ci < 2; ++ci)
#pragma unroll
    for (int k = 0; k < 3; ++k) {
      const float xv = xp[ci * 12 + t + k];
      a += w[(co * 2 + ci) * 3 + k] * xv;
      g += w[((co + 64) * 2 + ci) * 3 + k] * xv;
    }
  out[idx] = a * sigmoidf_(g);
}

__global__ void deg_kernel(const int* __restrict__ col, float* deg, int n_edges) {
  int e = blockIdx.x * blockDim.x + threadIdx.x;
  if (e < n_edges) atomicAdd(&deg[col[e]], 1.0f);
}

__global__ void dinv_kernel(const float* __restrict__ deg, float* dinv, int n_nodes) {
  int i = blockIdx.x * blockDim.x + threadIdx.x;
  if (i < n_nodes) dinv[i] = rsqrtf(deg[i] + 1.0f);  // +1 self loop
}

// one thread per (edge, feature element)
template<int CT>
__global__ void gcn_scatter(const int* __restrict__ row, const int* __restrict__ col,
                            const float* __restrict__ dinv, const float* __restrict__ xw,
                            float* agg, int n_edges) {
  long long idx = blockIdx.x * (long long)blockDim.x + threadIdx.x;
  long long tot = (long long)n_edges * CT;
  if (idx >= tot) return;
  const int e = (int)(idx / CT);
  const int i = (int)(idx % CT);
  const int r = row[e], c = col[e];
  const float coef = dinv[r] * dinv[c];
  atomicAdd(&agg[(long long)c * CT + i], xw[(long long)r * CT + i] * coef);
}

// out may alias agg (in-place): one-to-one read-then-write
template<int CT, int C>
__global__ void gcn_finalize(const float* agg, const float* __restrict__ xw,
                             const float* __restrict__ dinv,
                             const float* __restrict__ bias,
                             float* out, int n_nodes) {
  constexpr int T = CT / C;
  long long idx = blockIdx.x * (long long)blockDim.x + threadIdx.x;
  long long tot = (long long)n_nodes * CT;
  if (idx >= tot) return;
  const long long n = idx / CT;
  const int i = (int)(idx % CT);
  const int c = i / T;
  const float d = dinv[n];
  const float v = agg[idx] + xw[idx] * d * d + bias[c];
  out[idx] = fmaxf(v, 0.0f);
}

__global__ void gap_sum(const float* __restrict__ h, const int* __restrict__ batch,
                        float* pooled, int n_nodes) {
  long long idx = blockIdx.x * (long long)blockDim.x + threadIdx.x;
  long long tot = (long long)n_nodes * 96;
  if (idx >= tot) return;
  const long long n = idx / 96;
  const int i = (int)(idx % 96);
  atomicAdd(&pooled[batch[n] * 96 + i], h[idx]);
}

__global__ void gap_cnt(const int* __restrict__ batch, float* cnt, int n_nodes) {
  int n = blockIdx.x * blockDim.x + threadIdx.x;
  if (n < n_nodes) atomicAdd(&cnt[batch[n]], 1.0f);
}

__global__ void gap_div(float* pooled, const float* __restrict__ cnt) {
  int idx = blockIdx.x * blockDim.x + threadIdx.x;
  if (idx < 16 * 96) pooled[idx] /= fmaxf(cnt[idx / 96], 1.0f);
}

// final conv_glu: pooled (16,16,6), w (128,16,3) -> out (16,64,4). Tiny.
__global__ void conv4_kernel(const float* __restrict__ p,
                             const float* __restrict__ w,
                             const float* __restrict__ b,
                             float* __restrict__ out) {
  int idx = blockIdx.x * blockDim.x + threadIdx.x;
  if (idx >= 16 * 64 * 4) return;
  const int t  = idx % 4;
  const int co = (idx / 4) % 64;
  const int gi = idx / 256;
  float a = b[co], g = b[co + 64];
  for (int ci = 0; ci < 16; ++ci)
#pragma unroll
    for (int k = 0; k < 3; ++k) {
      const float xv = p[(gi * 16 + ci) * 6 + t + k];
      a += w[(co * 16 + ci) * 3 + k] * xv;
      g += w[((co + 64) * 16 + ci) * 3 + k] * xv;
    }
  out[idx] = a * sigmoidf_(g);
}

// ---------------------------------------------------------------------------

extern "C" void kernel_launch(void* const* d_in, const int* in_sizes, int n_in,
                              void* d_out, int out_size, void* d_ws, size_t ws_size,
                              hipStream_t stream)
{
  const float* x     = (const float*)d_in[0];
  const int*   eidx  = (const int*)  d_in[1];
  const int*   batch = (const int*)  d_in[2];
  const float* w_t1a = (const float*)d_in[3];
  const float* b_t1a = (const float*)d_in[4];
  const float* w_s1  = (const float*)d_in[5];
  const float* bb_s1 = (const float*)d_in[6];
  const float* w_t1b = (const float*)d_in[7];
  const float* b_t1b = (const float*)d_in[8];
  const float* w_t2a = (const float*)d_in[9];
  const float* b_t2a = (const float*)d_in[10];
  const float* w_s2  = (const float*)d_in[11];
  const float* bb_s2 = (const float*)d_in[12];
  const float* w_t2b = (const float*)d_in[13];
  const float* b_t2b = (const float*)d_in[14];

  const int n_nodes = in_sizes[2];
  const int n_edges = in_sizes[1] / 2;
  const int* row = eidx;
  const int* col = eidx + n_edges;

  // workspace bump allocator (with buffer reuse, ~240 MB total)
  char* ws = (char*)d_ws;
  size_t off = 0;
  auto alloc = [&](size_t bytes) -> float* {
    float* p = (float*)(ws + off);
    off += (bytes + 255) & ~(size_t)255;
    return p;
  };
  float* bufA   = alloc((size_t)n_nodes * 64 * 10 * 4); // h1, then h3 (N,64,8)
  float* bufB   = alloc((size_t)n_nodes * 16 * 10 * 4); // xw1, then xw2
  float* bufC   = alloc((size_t)n_nodes * 16 * 10 * 4); // agg1/h2, then agg2/h5
  float* bufD   = alloc((size_t)n_nodes * 64 * 6 * 4);  // h4
  float* deg    = alloc((size_t)n_nodes * 4);
  float* dinv   = alloc((size_t)n_nodes * 4);
  float* pooled = alloc((size_t)16 * 16 * 6 * 4);
  float* cnt    = alloc((size_t)16 * 4);
  (void)ws_size; (void)n_in; (void)out_size;

  auto cdiv = [](long long a, long long b) { return (unsigned)((a + b - 1) / b); };

  // 1) conv_glu t1a (scalar): x -> h1 (N,64,10)
  conv1_kernel<<<cdiv((long long)n_nodes * 640, 256), 256, 0, stream>>>(
      x, w_t1a, b_t1a, bufA, n_nodes);

  // 2) einsum1 (WMMA): h1 x w_s1 -> xw1 (N,16,10)
  conv_wmma_kernel<64, 1, 16, 10, false, true>
      <<<cdiv(((long long)n_nodes * 10 + 15) / 16, 4), 128, 0, stream>>>(
          bufA, w_s1, nullptr, bufB, n_nodes);

  // 3) degrees + dinv
  zero_f32<<<cdiv(n_nodes, 256), 256, 0, stream>>>(deg, n_nodes);
  deg_kernel<<<cdiv(n_edges, 256), 256, 0, stream>>>(col, deg, n_edges);
  dinv_kernel<<<cdiv(n_nodes, 256), 256, 0, stream>>>(deg, dinv, n_nodes);

  // 4) GCN #1: scatter + self-loop + bias + relu -> h2 (N,16,10) in bufC
  zero_f32<<<cdiv((long long)n_nodes * 160, 256), 256, 0, stream>>>(
      bufC, (long long)n_nodes * 160);
  gcn_scatter<160><<<cdiv((long long)n_edges * 160, 256), 256, 0, stream>>>(
      row, col, dinv, bufB, bufC, n_edges);
  gcn_finalize<160, 16><<<cdiv((long long)n_nodes * 160, 256), 256, 0, stream>>>(
      bufC, bufB, dinv, bb_s1, bufC, n_nodes);

  // 5) conv_glu t1b (WMMA, K=48): h2 -> h3 (N,64,8) in bufA
  conv_wmma_kernel<16, 3, 128, 10, true, false>
      <<<cdiv(((long long)n_nodes * 8 + 15) / 16, 4), 128, 0, stream>>>(
          bufC, w_t1b, b_t1b, bufA, n_nodes);

  // 6) conv_glu t2a (WMMA, K=192): h3 -> h4 (N,64,6) in bufD
  conv_wmma_kernel<64, 3, 128, 8, true, false>
      <<<cdiv(((long long)n_nodes * 6 + 15) / 16, 4), 128, 0, stream>>>(
          bufA, w_t2a, b_t2a, bufD, n_nodes);

  // 7) einsum2 (WMMA): h4 x w_s2 -> xw2 (N,16,6) in bufB
  conv_wmma_kernel<64, 1, 16, 6, false, true>
      <<<cdiv(((long long)n_nodes * 6 + 15) / 16, 4), 128, 0, stream>>>(
          bufD, w_s2, nullptr, bufB, n_nodes);

  // 8) GCN #2 -> h5 (N,16,6) in bufC
  zero_f32<<<cdiv((long long)n_nodes * 96, 256), 256, 0, stream>>>(
      bufC, (long long)n_nodes * 96);
  gcn_scatter<96><<<cdiv((long long)n_edges * 96, 256), 256, 0, stream>>>(
      row, col, dinv, bufB, bufC, n_edges);
  gcn_finalize<96, 16><<<cdiv((long long)n_nodes * 96, 256), 256, 0, stream>>>(
      bufC, bufB, dinv, bb_s2, bufC, n_nodes);

  // 9) global mean pool -> pooled (16,16,6)
  zero_f32<<<cdiv(16 * 96, 256), 256, 0, stream>>>(pooled, 16 * 96);
  zero_f32<<<1, 32, 0, stream>>>(cnt, 16);
  gap_sum<<<cdiv((long long)n_nodes * 96, 256), 256, 0, stream>>>(
      bufC, batch, pooled, n_nodes);
  gap_cnt<<<cdiv(n_nodes, 256), 256, 0, stream>>>(batch, cnt, n_nodes);
  gap_div<<<cdiv(16 * 96, 256), 256, 0, stream>>>(pooled, cnt);

  // 10) conv_glu t2b (scalar, tiny): pooled -> out (16,64,4)
  conv4_kernel<<<cdiv(16 * 64 * 4, 256), 256, 0, stream>>>(
      pooled, w_t2b, b_t2b, (float*)d_out);
}
